// TSCluster_48292612276528
// MI455X (gfx1250) — matrix operand order repaired and checked by reading, not compiled, not run on hardware
//
#include <hip/hip_runtime.h>
#include <hip/hip_bf16.h>

typedef __attribute__((ext_vector_type(16))) _Float16 v16h;
typedef __attribute__((ext_vector_type(8)))  _Float16 v8h;
typedef __attribute__((ext_vector_type(4)))  _Float16 v4h;
typedef __attribute__((ext_vector_type(8)))  float    v8f;

#define EPSC 1e-10f

// ---------------------------------------------------------------------------
// WMMA GEMM: C[M,Nout] = epilogue(A[M,Kin] @ W[Nrows,Kin]^T + bias)
// A f32 global, W f16 global (rows padded to multiple of 64, zero-filled).
// Requires: M % 128 == 0, Kin % 32 == 0.
// FLAGS (compile time): 1=bias, 2=relu, 4=residual. GUARDN: guard col<Nout.
// Block 256 thr = 8 waves (4 M x 2 N); tile 128x64, BK=32; wave = 32x32 out.
// Double-buffered LDS, one barrier per K-step, global loads overlap WMMA.
// ---------------------------------------------------------------------------
#define GBM 128
#define GBN 64
#define GBK 32

template<int FLAGS, bool GUARDN>
__global__ __launch_bounds__(256) void ts_gemm_wmma(
    const float* __restrict__ A, const _Float16* __restrict__ W,
    const float* __restrict__ bias, const float* __restrict__ resid,
    float* __restrict__ C, int M, int Nout, int Kin)
{
    __shared__ _Float16 As[2][GBM][GBK + 8];
    __shared__ _Float16 Bs[2][GBN][GBK + 8];

    const int tid  = threadIdx.x;
    const int wave = tid >> 5;
    const int lane = tid & 31;
    const int wm   = wave & 3;
    const int wn   = wave >> 2;
    const int m0   = blockIdx.x * GBM;
    const int n0   = blockIdx.y * GBN;
    const int l15  = lane & 15;
    const int hi   = lane >> 4;
    const int koffA = hi ? 8 : 0;   // A frag: lanes>=16 hold K 8..15 / 24..31
    const int koffB = hi ? 8 : 0;   // B frag: lanes>=16 hold K 8..15 / 24..31

    // staging assignments
    const int ar  = tid >> 1;                 // A row 0..127
    const int akb = (tid & 1) * 16;           // A k-base 0 or 16
    const float4* asrc = (const float4*)(A + (size_t)(m0 + ar) * Kin + akb);
    const int br  = tid >> 2;                 // B row 0..63
    const int bkb = (tid & 3) * 8;            // B k-base 0/8/16/24
    const _Float16* bsrc = W + (size_t)(n0 + br) * Kin + bkb;

    const int nsteps = Kin >> 5;
    v8f acc[2][2] = {};

    // prologue: stage tile 0 into buffer 0
    {
        #pragma unroll
        for (int i = 0; i < 4; ++i) {
            const float4 f = asrc[i];
            v4h h = { (_Float16)f.x, (_Float16)f.y, (_Float16)f.z, (_Float16)f.w };
            *(v4h*)&As[0][ar][akb + 4 * i] = h;
        }
        *(v8h*)&Bs[0][br][bkb] = *(const v8h*)bsrc;
    }
    __syncthreads();

    for (int step = 0; step < nsteps; ++step) {
        const int cur = step & 1;
        const bool has_next = (step + 1) < nsteps;

        // issue global loads for next tile early (hide under WMMA)
        float4 af[4];
        v8h bfv;
        if (has_next) {
            asrc += 8;          // +32 floats
            bsrc += GBK;
            #pragma unroll
            for (int i = 0; i < 4; ++i) af[i] = asrc[i];
            bfv = *(const v8h*)bsrc;
        }

        // fragments from current buffer
        v16h afr[2], bfr[2];
        #pragma unroll
        for (int i = 0; i < 2; ++i) {
            const int row = wm * 32 + i * 16 + l15;
            v8h lo = *(const v8h*)&As[cur][row][koffA];
            v8h hh = *(const v8h*)&As[cur][row][16 + koffA];
            afr[i] = __builtin_shufflevector(lo, hh, 0,1,2,3,4,5,6,7,8,9,10,11,12,13,14,15);
        }
        #pragma unroll
        for (int j = 0; j < 2; ++j) {
            const int col = wn * 32 + j * 16 + l15;
            v8h lo = *(const v8h*)&Bs[cur][col][koffB];
            v8h hh = *(const v8h*)&Bs[cur][col][koffB + 16];
            bfr[j] = __builtin_shufflevector(lo, hh, 0,1,2,3,4,5,6,7,8,9,10,11,12,13,14,15);
        }
        #pragma unroll
        for (int i = 0; i < 2; ++i)
            #pragma unroll
            for (int j = 0; j < 2; ++j)
                acc[i][j] = __builtin_amdgcn_wmma_f32_16x16x32_f16(
                    false, afr[i], false, bfr[j], (short)0, acc[i][j], false, false);

        // stage next tile into alternate buffer; single barrier per step
        if (has_next) {
            const int nxt = cur ^ 1;
            #pragma unroll
            for (int i = 0; i < 4; ++i) {
                v4h h = { (_Float16)af[i].x, (_Float16)af[i].y,
                          (_Float16)af[i].z, (_Float16)af[i].w };
                *(v4h*)&As[nxt][ar][akb + 4 * i] = h;
            }
            *(v8h*)&Bs[nxt][br][bkb] = bfv;
            __syncthreads();
        }
    }
    (void)M;

    // straight-line epilogue (FLAGS compile-time)
    #pragma unroll
    for (int j = 0; j < 2; ++j) {
        const int col = n0 + wn * 32 + j * 16 + l15;
        if (GUARDN && col >= Nout) continue;
        const float bv = (FLAGS & 1) ? bias[col] : 0.0f;
        #pragma unroll
        for (int i = 0; i < 2; ++i) {
            const int row0 = m0 + wm * 32 + i * 16 + 8 * hi;
            float* cptr = C + (size_t)row0 * Nout + col;
            const float* rptr = (FLAGS & 4) ? (resid + (size_t)row0 * Nout + col) : nullptr;
            #pragma unroll
            for (int r = 0; r < 8; ++r) {
                float v = acc[i][j][r] + bv;
                if (FLAGS & 2) v = v > 0.0f ? v : 0.0f;
                if (FLAGS & 4) v += rptr[(size_t)r * Nout];
                cptr[(size_t)r * Nout] = v;
            }
        }
    }
}

// ---------------------------------------------------------------------------
// Fused attention per (batch, head): q,k,v [64,32] -> o [64,32]
// Q@K^T via WMMA (K=32), f32 softmax, P@V via WMMA (K=64 in 2 steps).
// ---------------------------------------------------------------------------
__global__ __launch_bounds__(128) void ts_attn_wmma(
    const float* __restrict__ qkv,   // [Mc, 768]
    float* __restrict__ out)         // [Mc, 256]
{
    __shared__ _Float16 Qh[64][40];
    __shared__ _Float16 Kh[64][40];
    __shared__ _Float16 Vt[32][72];
    __shared__ float    Sf[64][68];
    __shared__ _Float16 Ph[64][72];

    const int blk = blockIdx.x;
    const int b   = blk >> 3;
    const int h   = blk & 7;
    const int tid = threadIdx.x;
    const int wave = tid >> 5, lane = tid & 31;
    const size_t base = (size_t)b * 64 * 768 + (size_t)h * 32;

    // stage Q,K (vectorized)
    {
        const int r  = tid >> 1;
        const int cb = (tid & 1) * 16;
        const float4* q = (const float4*)(qkv + base + (size_t)r * 768 + cb);
        #pragma unroll
        for (int i = 0; i < 4; ++i) {
            const float4 fq = q[i];
            const float4 fk = q[64 + i];
            v4h hq = { (_Float16)fq.x, (_Float16)fq.y, (_Float16)fq.z, (_Float16)fq.w };
            v4h hk = { (_Float16)fk.x, (_Float16)fk.y, (_Float16)fk.z, (_Float16)fk.w };
            *(v4h*)&Qh[r][cb + 4 * i] = hq;
            *(v4h*)&Kh[r][cb + 4 * i] = hk;
        }
    }
    // stage V transposed: lane = head-dim (coalesced), pack 4 tokens per store
    {
        const int t0 = wave * 16;
        const float* v = qkv + base + 512 + lane;
        #pragma unroll
        for (int tb = 0; tb < 16; tb += 4) {
            v4h hv;
            #pragma unroll
            for (int j = 0; j < 4; ++j)
                hv[j] = (_Float16)v[(size_t)(t0 + tb + j) * 768];
            *(v4h*)&Vt[lane][t0 + tb] = hv;
        }
    }
    __syncthreads();

    const int l15 = lane & 15, hi = lane >> 4;
    const int koffA = hi ? 8 : 0;
    const int koffB = hi ? 8 : 0;
    const float scale = 0.17677669529663687f;  // 1/sqrt(32)

    // scores S = Q K^T
    v16h aq;
    {
        const int row = wave * 16 + l15;
        v8h lo = *(const v8h*)&Qh[row][koffA];
        v8h hh = *(const v8h*)&Qh[row][16 + koffA];
        aq = __builtin_shufflevector(lo, hh, 0,1,2,3,4,5,6,7,8,9,10,11,12,13,14,15);
    }
    #pragma unroll
    for (int ct = 0; ct < 4; ++ct) {
        const int col = ct * 16 + l15;
        v8h lo = *(const v8h*)&Kh[col][koffB];
        v8h hh = *(const v8h*)&Kh[col][koffB + 16];
        v16h bk = __builtin_shufflevector(lo, hh, 0,1,2,3,4,5,6,7,8,9,10,11,12,13,14,15);
        v8f s = {};
        s = __builtin_amdgcn_wmma_f32_16x16x32_f16(false, aq, false, bk, (short)0, s, false, false);
        #pragma unroll
        for (int r = 0; r < 8; ++r)
            Sf[wave * 16 + r + 8 * hi][ct * 16 + l15] = s[r] * scale;
    }
    __syncthreads();

    // row softmax (64 rows, one thread each)
    if (tid < 64) {
        float mx = -1e30f;
        for (int j = 0; j < 64; ++j) mx = fmaxf(mx, Sf[tid][j]);
        float sum = 0.0f;
        for (int j = 0; j < 64; ++j) { float e = __expf(Sf[tid][j] - mx); Sf[tid][j] = e; sum += e; }
        const float inv = 1.0f / sum;
        for (int j = 0; j < 64; ++j) Ph[tid][j] = (_Float16)(Sf[tid][j] * inv);
    }
    __syncthreads();

    // o = P @ V
    #pragma unroll
    for (int ctv = 0; ctv < 2; ++ctv) {
        v8f o = {};
        #pragma unroll
        for (int ks = 0; ks < 2; ++ks) {
            const int row = wave * 16 + l15;
            v8h alo = *(const v8h*)&Ph[row][ks * 32 + koffA];
            v8h ahh = *(const v8h*)&Ph[row][ks * 32 + 16 + koffA];
            v16h ap = __builtin_shufflevector(alo, ahh, 0,1,2,3,4,5,6,7,8,9,10,11,12,13,14,15);
            const int col = ctv * 16 + l15;
            v8h blo = *(const v8h*)&Vt[col][ks * 32 + koffB];
            v8h bhh = *(const v8h*)&Vt[col][ks * 32 + koffB + 16];
            v16h bv = __builtin_shufflevector(blo, bhh, 0,1,2,3,4,5,6,7,8,9,10,11,12,13,14,15);
            o = __builtin_amdgcn_wmma_f32_16x16x32_f16(false, ap, false, bv, (short)0, o, false, false);
        }
        #pragma unroll
        for (int r = 0; r < 8; ++r) {
            const int tok = wave * 16 + r + 8 * hi;
            out[((size_t)b * 64 + tok) * 256 + h * 32 + ctv * 16 + l15] = o[r];
        }
    }
}

// ---------------------------------------------------------------------------
// LayerNorm in place over rows of length 256. One wave per row, 8 rows/block.
// ---------------------------------------------------------------------------
__global__ __launch_bounds__(256) void ts_layernorm(
    float* __restrict__ x, const float* __restrict__ g,
    const float* __restrict__ b, int M)
{
    const int wave = threadIdx.x >> 5, lane = threadIdx.x & 31;
    const int row = blockIdx.x * 8 + wave;
    if (row >= M) return;
    float* p = x + (size_t)row * 256;
    float vals[8];
    float s = 0.0f;
    #pragma unroll
    for (int i = 0; i < 8; ++i) { vals[i] = p[lane + 32 * i]; s += vals[i]; }
    #pragma unroll
    for (int off = 16; off; off >>= 1) s += __shfl_xor(s, off, 32);
    const float mean = s * (1.0f / 256.0f);
    float vs = 0.0f;
    #pragma unroll
    for (int i = 0; i < 8; ++i) { const float d = vals[i] - mean; vs += d * d; }
    #pragma unroll
    for (int off = 16; off; off >>= 1) vs += __shfl_xor(vs, off, 32);
    const float inv = rsqrtf(vs * (1.0f / 256.0f) + 1e-5f);
    #pragma unroll
    for (int i = 0; i < 8; ++i) {
        const int c = lane + 32 * i;
        p[c] = (vals[i] - mean) * inv * g[c] + b[c];
    }
}

// f32 -> f16 conversion (weights)
__global__ void ts_cvt_f16(const float* __restrict__ s, _Float16* __restrict__ d, int n)
{
    const int i = blockIdx.x * blockDim.x + threadIdx.x;
    if (i < n) d[i] = (_Float16)s[i];
}

// tanh(mus) -> f16 [64,256] (rows 32..63 zero pad) and ||tanh(mu_k)||^2
__global__ __launch_bounds__(64) void ts_mt(
    const float* __restrict__ mus, _Float16* __restrict__ mt, float* __restrict__ mt2)
{
    const int k = threadIdx.x;
    if (k < 32) {
        float s = 0.0f;
        for (int d = 0; d < 256; ++d) {
            const float t = tanhf(mus[k * 256 + d]);
            mt[k * 256 + d] = (_Float16)t;
            s += t * t;
        }
        mt2[k] = s;
    } else {
        for (int d = 0; d < 256; ++d) mt[k * 256 + d] = (_Float16)0.0f;
    }
}

// softmax over 32: block 0 -> pi, blocks 1..32 -> rows of A
__global__ __launch_bounds__(32) void ts_softmax32(
    const float* __restrict__ il, const float* __restrict__ tl,
    float* __restrict__ pi, float* __restrict__ Amat)
{
    const int lane = threadIdx.x;
    const float* src; float* dst;
    if (blockIdx.x == 0) { src = il; dst = pi; }
    else { src = tl + (blockIdx.x - 1) * 32; dst = Amat + (blockIdx.x - 1) * 32; }
    float v = src[lane];
    float mx = v;
    #pragma unroll
    for (int off = 16; off; off >>= 1) mx = fmaxf(mx, __shfl_xor(mx, off, 32));
    const float e = __expf(v - mx);
    float sum = e;
    #pragma unroll
    for (int off = 16; off; off >>= 1) sum += __shfl_xor(sum, off, 32);
    dst[lane] = e / sum;
}

// Cluster probs in place on dot products. One wave per row.
__global__ __launch_bounds__(256) void ts_cluster_prob(
    const float* __restrict__ emb, float* __restrict__ dp,
    const float* __restrict__ mt2, int Mtot)
{
    const int wave = threadIdx.x >> 5, lane = threadIdx.x & 31;
    const int row = blockIdx.x * 8 + wave;
    if (row >= Mtot) return;
    const float* e = emb + (size_t)row * 256;
    float s = 0.0f;
    #pragma unroll
    for (int i = 0; i < 8; ++i) { const float v = e[lane + 32 * i]; s += v * v; }
    #pragma unroll
    for (int off = 16; off; off >>= 1) s += __shfl_xor(s, off, 32);
    const float dot = dp[(size_t)row * 32 + lane];
    const float d2 = s - 2.0f * dot + mt2[lane];
    const float ex = __expf(-d2 * 0.1f);
    float tot = ex;
    #pragma unroll
    for (int off = 16; off; off >>= 1) tot += __shfl_xor(tot, off, 32);
    dp[(size_t)row * 32 + lane] = ex / tot;
}

// HMM forward recursion per batch (32 lanes = 32 states), in place.
__global__ __launch_bounds__(32) void ts_hmm(
    float* __restrict__ cp, const float* __restrict__ pi,
    const float* __restrict__ Amat, float* __restrict__ entpart)
{
    __shared__ float As[32][33];
    const int b = blockIdx.x, k = threadIdx.x;
    for (int j = 0; j < 32; ++j) As[j][k] = Amat[j * 32 + k];
    __syncthreads();

    float* row = cp + (size_t)b * 64 * 32;
    float ent = 0.0f;

    float val = __logf(pi[k] + EPSC) + __logf(row[k] + EPSC);
    float mx = val;
    #pragma unroll
    for (int off = 16; off; off >>= 1) mx = fmaxf(mx, __shfl_xor(mx, off, 32));
    float e = __expf(val - mx);
    float sm = e;
    #pragma unroll
    for (int off = 16; off; off >>= 1) sm += __shfl_xor(sm, off, 32);
    float lt = e / sm;
    row[k] = lt;
    {
        float t = lt * __logf(lt + EPSC), ts = t;
        #pragma unroll
        for (int off = 16; off; off >>= 1) ts += __shfl_xor(ts, off, 32);
        ent += ts;
    }

    for (int n = 1; n < 64; ++n) {
        float accv = 0.0f;
        for (int j = 0; j < 32; ++j) accv += __shfl(lt, j, 32) * As[j][k];
        float* r = row + n * 32;
        val = __logf(accv + EPSC) + __logf(r[k] + EPSC);
        mx = val;
        #pragma unroll
        for (int off = 16; off; off >>= 1) mx = fmaxf(mx, __shfl_xor(mx, off, 32));
        e = __expf(val - mx);
        sm = e;
        #pragma unroll
        for (int off = 16; off; off >>= 1) sm += __shfl_xor(sm, off, 32);
        lt = e / sm;
        r[k] = lt;
        float t = lt * __logf(lt + EPSC), ts = t;
        #pragma unroll
        for (int off = 16; off; off >>= 1) ts += __shfl_xor(ts, off, 32);
        ent += ts;
    }
    if (k == 0) entpart[b] = ent;
}

// fixed-order deterministic reduction of 2048 per-batch entropy sums
__global__ __launch_bounds__(256) void ts_ent_final(
    const float* __restrict__ entpart, float* __restrict__ out)
{
    __shared__ float sh[256];
    float s = 0.0f;
    for (int i = threadIdx.x; i < 2048; i += 256) s += entpart[i];
    sh[threadIdx.x] = s;
    __syncthreads();
    for (int off = 128; off; off >>= 1) {
        if (threadIdx.x < off) sh[threadIdx.x] += sh[threadIdx.x + off];
        __syncthreads();
    }
    if (threadIdx.x == 0) out[0] = -sh[0] / (2048.0f * 64.0f);
}

// ---------------------------------------------------------------------------
// Host driver
// ---------------------------------------------------------------------------
extern "C" void kernel_launch(void* const* d_in, const int* in_sizes, int n_in,
                              void* d_out, int out_size, void* d_ws, size_t ws_size,
                              hipStream_t stream)
{
    const int Bc = 2048, Nn = 64, Ll = 2, DFFc = 1024;
    const int Mtot = Bc * Nn;              // 131072
    const int CB = 128;                    // batches per chunk
    const int NCHUNK = Bc / CB;            // 16
    const int Mc = CB * Nn;                // 8192

    const float* input = (const float*)d_in[0];
    const float* il    = (const float*)d_in[1];
    const float* tl    = (const float*)d_in[2];
    const float* Wqkv  = (const float*)d_in[3];
    const float* bqkv  = (const float*)d_in[4];
    const float* Wo    = (const float*)d_in[5];
    const float* bo    = (const float*)d_in[6];
    const float* ln1g  = (const float*)d_in[7];
    const float* ln1b  = (const float*)d_in[8];
    const float* ln2g  = (const float*)d_in[9];
    const float* ln2b  = (const float*)d_in[10];
    const float* W1    = (const float*)d_in[11];
    const float* b1    = (const float*)d_in[12];
    const float* W2    = (const float*)d_in[13];
    const float* b2    = (const float*)d_in[14];
    const float* mus   = (const float*)d_in[15];

    float* out   = (float*)d_out;
    float* cp    = out;                                  // [B,N,32]
    float* emb   = out + (size_t)Mtot * 32;              // [B,N,256] activations
    float* piO   = out + (size_t)Mtot * 32 + (size_t)Mtot * 256;
    float* AO    = piO + 32;
    float* entO  = AO + 1024;

    // workspace layout
    char* ws = (char*)d_ws;
    size_t o = 0;
    _Float16* wqkv16 = (_Float16*)(ws + o); o += (size_t)Ll * 768 * 256 * 2;
    _Float16* wo16   = (_Float16*)(ws + o); o += (size_t)Ll * 256 * 256 * 2;
    _Float16* w116   = (_Float16*)(ws + o); o += (size_t)Ll * 1024 * 256 * 2;
    _Float16* w216   = (_Float16*)(ws + o); o += (size_t)Ll * 256 * 1024 * 2;
    _Float16* mt16   = (_Float16*)(ws + o); o += (size_t)64 * 256 * 2;   // padded to 64 rows
    float* mt2ws     = (float*)(ws + o);    o += 256;
    float* entpart   = (float*)(ws + o);    o += 2048 * 4;
    float* qkvbuf    = (float*)(ws + o);    o += (size_t)Mc * 768 * 4;
    float* attnbuf   = (float*)(ws + o);    o += (size_t)Mc * 256 * 4;
    float* ffnbuf    = (float*)(ws + o);    o += (size_t)Mc * 1024 * 4;
    (void)ws_size; (void)n_in; (void)in_sizes; (void)out_size;

    // residual stream lives in the d_out emb region
    hipMemcpyAsync(emb, input, (size_t)Mtot * 256 * sizeof(float),
                   hipMemcpyDeviceToDevice, stream);

    // weight conversion
    {
        int n;
        n = Ll * 768 * 256;  ts_cvt_f16<<<(n + 255) / 256, 256, 0, stream>>>(Wqkv, wqkv16, n);
        n = Ll * 256 * 256;  ts_cvt_f16<<<(n + 255) / 256, 256, 0, stream>>>(Wo,   wo16,   n);
        n = Ll * 1024 * 256; ts_cvt_f16<<<(n + 255) / 256, 256, 0, stream>>>(W1,   w116,   n);
        n = Ll * 256 * 1024; ts_cvt_f16<<<(n + 255) / 256, 256, 0, stream>>>(W2,   w216,   n);
    }
    ts_mt<<<1, 64, 0, stream>>>(mus, mt16, mt2ws);
    ts_softmax32<<<33, 32, 0, stream>>>(il, tl, piO, AO);

    // encoder
    for (int l = 0; l < Ll; ++l) {
        const _Float16* wq = wqkv16 + (size_t)l * 768 * 256;
        const _Float16* wo = wo16   + (size_t)l * 256 * 256;
        const _Float16* w1 = w116   + (size_t)l * 1024 * 256;
        const _Float16* w2 = w216   + (size_t)l * 256 * 1024;
        for (int c = 0; c < NCHUNK; ++c) {
            float* xc = emb + (size_t)c * Mc * 256;
            ts_gemm_wmma<1, false><<<dim3(Mc / GBM, 768 / GBN), 256, 0, stream>>>(
                xc, wq, bqkv + l * 768, nullptr, qkvbuf, Mc, 768, 256);
            ts_attn_wmma<<<CB * 8, 128, 0, stream>>>(qkvbuf, attnbuf);
            ts_gemm_wmma<1 | 4, false><<<dim3(Mc / GBM, 256 / GBN), 256, 0, stream>>>(
                attnbuf, wo, bo + l * 256, xc, xc, Mc, 256, 256);
            ts_layernorm<<<Mc / 8, 256, 0, stream>>>(xc, ln1g + l * 256, ln1b + l * 256, Mc);
            ts_gemm_wmma<1 | 2, false><<<dim3(Mc / GBM, DFFc / GBN), 256, 0, stream>>>(
                xc, w1, b1 + l * 1024, nullptr, ffnbuf, Mc, DFFc, 256);
            ts_gemm_wmma<1 | 4, false><<<dim3(Mc / GBM, 256 / GBN), 256, 0, stream>>>(
                ffnbuf, w2, b2 + l * 256, xc, xc, Mc, 256, DFFc);
            ts_layernorm<<<Mc / 8, 256, 0, stream>>>(xc, ln2g + l * 256, ln2b + l * 256, Mc);
        }
    }

    // cluster emission: dot = emb @ tanh(mu)^T -> cp, then probs in place
    ts_gemm_wmma<0, true><<<dim3(Mtot / GBM, 1), 256, 0, stream>>>(
        emb, mt16, nullptr, nullptr, cp, Mtot, 32, 256);
    ts_cluster_prob<<<Mtot / 8, 256, 0, stream>>>(emb, cp, mt2ws, Mtot);

    // HMM forward recursion + entropy
    ts_hmm<<<Bc, 32, 0, stream>>>(cp, piO, AO, entpart);
    ts_ent_final<<<1, 256, 0, stream>>>(entpart, entO);
}
